// RegionProposalNetwork_10728828305965
// MI455X (gfx1250) — matrix operand order repaired and checked by reading, not compiled, not run on hardware
//
#include <hip/hip_runtime.h>
#include <hip/hip_bf16.h>

// ---------------------------------------------------------------------------
// RPN pipeline for MI455X (gfx1250, wave32, WMMA).
// Heavy op: 3x3 conv 512->512 over 128x128 == implicit GEMM
//   M=16384 (pixels), K=4608 (512 ic x 9 taps), N=512  => 77.3 GFLOP.
// Conv: block = 32 pixels x 512 outputs, A-halo staged in LDS (102KB of the
// 320KB/WGP) via GLOBAL_LOAD_ASYNC_TO_LDS_B128 (ASYNCcnt-tracked DMA, no VGPR
// landing buffers). Each wave holds 2x4 = 8 16x16 f32 accumulators
// -> 8 WMMAs per A-fragment pair, branch-free K loop fed by ds_load_b128.
// ---------------------------------------------------------------------------

typedef __bf16 bf16;
typedef __attribute__((ext_vector_type(16))) __bf16 v16bf;
typedef __attribute__((ext_vector_type(8)))  __bf16 v8bf;
typedef __attribute__((ext_vector_type(8)))  float  v8f;

#if defined(__has_builtin)
#if __has_builtin(__builtin_amdgcn_global_load_async_to_lds_b128)
#define HAVE_ASYNC_LDS 1
// exact parameter type per hipcc diagnostic: int __vector_size__(16), AS1/AS3
typedef int async_b128_t __attribute__((__vector_size__(16)));
typedef __attribute__((address_space(1))) async_b128_t* async_gptr_t;
typedef __attribute__((address_space(3))) async_b128_t* async_lptr_t;
#endif
#endif

static constexpr int IMG   = 128;         // feature map H = W
static constexpr int HWPIX = IMG * IMG;   // 16384
static constexpr int CIN   = 512;
static constexpr int COUT  = 512;
static constexpr int NPRE  = 6000;
static constexpr int NPOST = 300;

// conv tiling
static constexpr int MPIX   = 32;                   // pixels per block (one row span)
static constexpr int HALOW  = MPIX + 2;             // 34 halo columns
static constexpr int SXELEM = 3 * HALOW * CIN;      // 52224 bf16 = 104448 B LDS

// ---------------------------------------------------------------------------
// 1) x: NCHW f32 -> HWC bf16   (A-matrix K-contiguous per pixel)
// ---------------------------------------------------------------------------
__global__ void transpose_x_kernel(const float* __restrict__ x, bf16* __restrict__ xb) {
  int idx = blockIdx.x * 256 + threadIdx.x;       // 512*16384 total
  int c   = idx & (CIN - 1);
  int pix = idx >> 9;
  xb[idx] = (bf16)x[c * HWPIX + pix];
}

// ---------------------------------------------------------------------------
// 2) conv1_w: [oc][ic][kh][kw] f32 -> [tap][oc][ic] bf16 (B frag = 1x32B load)
// ---------------------------------------------------------------------------
__global__ void prep_w_kernel(const float* __restrict__ wsrc, bf16* __restrict__ wb) {
  int idx = blockIdx.x * 256 + threadIdx.x;       // 9*512*512 total
  int ic  = idx & 511;
  int rem = idx >> 9;                             // tap*512 + oc
  int oc  = rem & 511;
  int tap = rem >> 9;                             // 0..8
  wb[idx] = (bf16)wsrc[(oc * CIN + ic) * 9 + tap];
}

// ---------------------------------------------------------------------------
// 3) Implicit-GEMM conv via WMMA, LDS-staged A halo (async copy when available).
//    Block: 256 thr (8 waves) -> 32 pixels x 512 oc.
//    Wave: 2 M-subtiles x 4 N-subtiles = 8 v8f accumulators.
// ---------------------------------------------------------------------------
__global__ __launch_bounds__(256) void conv_wmma_kernel(
    const bf16* __restrict__ xb,      // [16384][512]
    const bf16* __restrict__ wb,      // [9][512 oc][512 ic]
    const float* __restrict__ bias,   // [512]
    const int*  __restrict__ mask,    // [16384]
    float* __restrict__ hout)         // [16384][512]
{
  extern __shared__ char smem_raw[];
  bf16* sx = (bf16*)smem_raw;                   // [3][34][512]

  const int lane  = threadIdx.x & 31;
  const int wave  = threadIdx.x >> 5;
  const int m0    = blockIdx.x * MPIX;          // 32 consecutive pixels, same row
  const int hrow  = m0 >> 7;
  const int wcol0 = m0 & 127;
  const int hi = lane >> 4;
  const int ln = lane & 15;
  const int nbase = wave * 64;                  // this wave's 64 output channels

  // ---- stage A halo: rows hrow-1..hrow+1, cols wcol0-1..wcol0+32, all 512 ch
  for (int i = threadIdx.x * 8; i < SXELEM; i += 256 * 8) {
    int c = i & (CIN - 1);
    int p = (i >> 9) % HALOW;
    int r = (i >> 9) / HALOW;
    int gr = hrow + r - 1;
    int gc = wcol0 + p - 1;
    const bool inimg = (gr >= 0) && (gr < IMG) && (gc >= 0) && (gc < IMG);
#ifdef HAVE_ASYNC_LDS
    if (inimg) {
      // direct global->LDS DMA, tracked by ASYNCcnt (no VGPR landing buffer)
      __builtin_amdgcn_global_load_async_to_lds_b128(
          (async_gptr_t)(xb + ((gr * IMG + gc) * CIN + c)),
          (async_lptr_t)(sx + i), 0, 0);
    } else {
      *(v8bf*)(sx + i) = (v8bf){};              // zero-pad the border
    }
#else
    v8bf v = {};
    if (inimg) v = *(const v8bf*)(xb + ((gr * IMG + gc) * CIN + c));
    *(v8bf*)(sx + i) = v;
#endif
  }
#ifdef HAVE_ASYNC_LDS
#if __has_builtin(__builtin_amdgcn_s_wait_asynccnt)
  __builtin_amdgcn_s_wait_asynccnt(0);
#else
  asm volatile("s_wait_asynccnt 0x0" ::: "memory");
#endif
#endif
  __syncthreads();

  v8f acc[2][4] = {};

  for (int tap = 0; tap < 9; ++tap) {
    const int kh = tap / 3, kw = tap % 3;
    // A rows live in LDS: halo col = ms*16 + ln + kw, halo row = kh
    const bf16* a0 = sx + ((kh * HALOW + (ln + kw)) * CIN);
    const bf16* a1 = sx + ((kh * HALOW + (16 + ln + kw)) * CIN);
    const bf16* bbase = wb + (((size_t)tap * COUT + nbase + ln) * CIN);
    if (tap < 8) {  // pull next tap's weight block toward the WGP
      __builtin_prefetch(wb + (((size_t)(tap + 1) * COUT + nbase + ln) * CIN), 0, 0);
    }
    for (int c0 = 0; c0 < CIN; c0 += 32) {
      // A 16x32 bf16 lane layout: element e -> k = 16*hi + e (matches B)
      v8bf l0 = *(const v8bf*)(a0 + c0 + 16 * hi);
      v8bf h0 = *(const v8bf*)(a0 + c0 + 16 * hi + 8);
      v8bf l1 = *(const v8bf*)(a1 + c0 + 16 * hi);
      v8bf h1 = *(const v8bf*)(a1 + c0 + 16 * hi + 8);
      v16bf A0 = __builtin_shufflevector(l0, h0, 0,1,2,3,4,5,6,7,8,9,10,11,12,13,14,15);
      v16bf A1 = __builtin_shufflevector(l1, h1, 0,1,2,3,4,5,6,7,8,9,10,11,12,13,14,15);
      // B 32x16 bf16: lane N = ln, element e -> k = 16*hi + e (ic-contiguous)
      #pragma unroll
      for (int t = 0; t < 4; ++t) {
        v16bf B = *(const v16bf*)(bbase + (size_t)t * 16 * CIN + c0 + 16 * hi);
        acc[0][t] = __builtin_amdgcn_wmma_f32_16x16x32_bf16(false, A0, false, B,
                                                            (short)0, acc[0][t], false, false);
        acc[1][t] = __builtin_amdgcn_wmma_f32_16x16x32_bf16(false, A1, false, B,
                                                            (short)0, acc[1][t], false, false);
      }
    }
  }

  // C layout: VGPR r -> M = 8*hi + r, N = ln
  #pragma unroll
  for (int t = 0; t < 4; ++t) {
    const int n = nbase + t * 16 + ln;
    const float bn = bias[n];
    #pragma unroll
    for (int ms = 0; ms < 2; ++ms) {
      #pragma unroll
      for (int r = 0; r < 8; ++r) {
        int m = m0 + ms * 16 + 8 * hi + r;
        float v = acc[ms][t][r] + bn;
        v = v > 0.0f ? v : 0.0f;
        v *= (float)mask[m];
        hout[(size_t)m * COUT + n] = v;
      }
    }
  }
}

// ---------------------------------------------------------------------------
// 4) Heads: per pixel 12 dot products over 512 channels (weights staged in LDS),
//    softmax/argmax over 7 classes, sigmoid cos head.
// ---------------------------------------------------------------------------
__global__ __launch_bounds__(256) void heads_kernel(
    const float* __restrict__ h,
    const float* __restrict__ loc_w, const float* __restrict__ loc_b,
    const float* __restrict__ cls_w, const float* __restrict__ cls_b,
    const float* __restrict__ cos_w, const float* __restrict__ cos_b,
    float* __restrict__ out_locs, float* __restrict__ out_cls,
    float* __restrict__ out_cos,
    float* __restrict__ fg, float* __restrict__ ascale,
    float* __restrict__ acosA, int* __restrict__ keepm)
{
  __shared__ float s_loc[4 * 512];
  __shared__ float s_cls[7 * 512];
  __shared__ float s_cos[512];
  const int tid = threadIdx.x;
  for (int i = tid; i < 4 * 512; i += 256) s_loc[i] = loc_w[i];
  for (int i = tid; i < 7 * 512; i += 256) s_cls[i] = cls_w[i];
  for (int i = tid; i < 512; i += 256)     s_cos[i] = cos_w[i];
  __syncthreads();

  const int m = blockIdx.x * 256 + tid;
  float l[4] = {loc_b[0], loc_b[1], loc_b[2], loc_b[3]};
  float cv[7];
  for (int j = 0; j < 7; ++j) cv[j] = cls_b[j];
  float co = cos_b[0];
  const float* hr = h + (size_t)m * COUT;
  for (int c = 0; c < CIN; ++c) {
    float hv = hr[c];
    l[0] += hv * s_loc[0 * 512 + c];
    l[1] += hv * s_loc[1 * 512 + c];
    l[2] += hv * s_loc[2 * 512 + c];
    l[3] += hv * s_loc[3 * 512 + c];
    for (int j = 0; j < 7; ++j) cv[j] += hv * s_cls[j * 512 + c];
    co += hv * s_cos[c];
  }
  for (int o = 0; o < 4; ++o) out_locs[m * 4 + o] = l[o];
  for (int j = 0; j < 7; ++j) out_cls[m * 7 + j] = cv[j];

  // softmax + first-occurrence argmax
  float mx = cv[0]; int arg = 0;
  for (int j = 1; j < 7; ++j) if (cv[j] > mx) { mx = cv[j]; arg = j; }
  float sum = 0.f, e0 = 0.f;
  for (int j = 0; j < 7; ++j) {
    float e = expf(cv[j] - mx);
    sum += e;
    if (j == 0) e0 = e;
  }
  fg[m] = 1.0f - e0 / sum;

  float sig = 1.0f / (1.0f + expf(-co));
  out_cos[m] = sig;
  float ac = sig;
  if (ac == 0.0f) ac += 1e-5f;
  acosA[m] = ac;
  const float BASE_SCALES[7] = {16.f, 9.f, 14.f, 21.f, 33.f, 54.f, 93.f};
  ascale[m] = BASE_SCALES[arg];
  keepm[m] = (arg > 0) ? 1 : 0;
}

// ---------------------------------------------------------------------------
// 5) Anchors + loc2bbox + clip + validity -> roi + scores (-inf for invalid)
// ---------------------------------------------------------------------------
__global__ __launch_bounds__(256) void anchor_roi_kernel(
    const float* __restrict__ locs, const float* __restrict__ fg,
    const float* __restrict__ ascale, const float* __restrict__ acosA,
    const int* __restrict__ keepm,
    const int* __restrict__ imgh, const int* __restrict__ imgw,
    float* __restrict__ out_anchor, float* __restrict__ roi,
    float* __restrict__ scores)
{
  const int m = blockIdx.x * 256 + threadIdx.x;
  const float a0 = ascale[0];
  const float c0 = acosA[0];
  const float ah = a0 * sqrtf(1.0f / (c0 * c0) - 1.0f);   // hvec[0], shared
  const float aw = ascale[m];
  const int hy = m >> 7, wx = m & 127;
  const float ay = hy * 8.0f + 4.0f;
  const float ax = wx * 8.0f + 4.0f;
  out_anchor[m * 4 + 0] = ay - 0.5f * ah;
  out_anchor[m * 4 + 1] = ax - 0.5f * aw;
  out_anchor[m * 4 + 2] = ay + 0.5f * ah;
  out_anchor[m * 4 + 3] = ax + 0.5f * aw;

  const float l0 = locs[m * 4 + 0], l1 = locs[m * 4 + 1];
  const float l2 = locs[m * 4 + 2], l3 = locs[m * 4 + 3];
  const float ncy = l0 * ah + ay;
  const float ncx = l1 * aw + ax;
  const float nh = expf(l2) * ah;
  const float nw = expf(l3) * aw;
  const float IH = (float)imgh[0], IW = (float)imgw[0];
  float r0 = fminf(fmaxf(ncy - 0.5f * nh, 0.0f), IH);
  float r1 = fminf(fmaxf(ncx - 0.5f * nw, 0.0f), IW);
  float r2 = fminf(fmaxf(ncy + 0.5f * nh, 0.0f), IH);
  float r3 = fminf(fmaxf(ncx + 0.5f * nw, 0.0f), IW);
  roi[m * 4 + 0] = r0; roi[m * 4 + 1] = r1;
  roi[m * 4 + 2] = r2; roi[m * 4 + 3] = r3;
  const bool valid = (keepm[m] != 0) && ((r2 - r0) >= 16.0f) && ((r3 - r1) >= 16.0f);
  scores[m] = valid ? fg[m] : -__builtin_inff();
}

// ---------------------------------------------------------------------------
// 6) Top-6000: full bitonic sort of 16384 (score,idx) in 128KB LDS
//    (CDNA5 WGP has 320KB LDS -> whole problem in one workgroup).
// ---------------------------------------------------------------------------
__global__ __launch_bounds__(1024) void sort_topk_kernel(
    const float* __restrict__ scores, const float* __restrict__ roi,
    float* __restrict__ cand, float* __restrict__ tsc)
{
  extern __shared__ char smem[];
  float* ss = (float*)smem;
  int*   si = (int*)(smem + HWPIX * sizeof(float));
  const int tid = threadIdx.x;
  for (int i = tid; i < HWPIX; i += 1024) { ss[i] = scores[i]; si[i] = i; }
  __syncthreads();
  for (int k = 2; k <= HWPIX; k <<= 1) {
    for (int j = k >> 1; j > 0; j >>= 1) {
      for (int i = tid; i < HWPIX; i += 1024) {
        int ixj = i ^ j;
        if (ixj > i) {
          bool desc = ((i & k) == 0);            // descending overall
          float a = ss[i], b = ss[ixj];
          bool sw = desc ? (a < b) : (a > b);
          if (sw) {
            ss[i] = b; ss[ixj] = a;
            int t = si[i]; si[i] = si[ixj]; si[ixj] = t;
          }
        }
      }
      __syncthreads();
    }
  }
  for (int i = tid; i < NPRE; i += 1024) {
    float s = ss[i]; int idx = si[i];
    bool fin = (s > -__builtin_inff());
    tsc[i] = s;
    cand[i * 4 + 0] = fin ? roi[idx * 4 + 0] : 0.f;
    cand[i * 4 + 1] = fin ? roi[idx * 4 + 1] : 0.f;
    cand[i * 4 + 2] = fin ? roi[idx * 4 + 2] : 0.f;
    cand[i * 4 + 3] = fin ? roi[idx * 4 + 3] : 0.f;
  }
}

// ---------------------------------------------------------------------------
// 7) NMS: single 1024-thread block; 300 rounds of parallel argmax + suppress.
// ---------------------------------------------------------------------------
__global__ __launch_bounds__(1024) void nms_kernel(
    const float* __restrict__ cand, const float* __restrict__ tsc,
    int* __restrict__ live, float* __restrict__ areas,
    float* __restrict__ out_rois, float* __restrict__ out_idx,
    float* __restrict__ out_scores, float* __restrict__ out_old)
{
  __shared__ float rs[1024];
  __shared__ int   ri[1024];
  __shared__ float bb[5];
  __shared__ int   keep_i[NPOST];
  __shared__ int   keep_v[NPOST];
  const int tid = threadIdx.x;

  for (int i = tid; i < NPRE; i += 1024) {
    float y0 = cand[i*4+0], x0 = cand[i*4+1], y1 = cand[i*4+2], x1 = cand[i*4+3];
    areas[i] = (y1 - y0) * (x1 - x0);
    live[i] = 1;
    out_old[i*4+0] = y0; out_old[i*4+1] = x0;
    out_old[i*4+2] = y1; out_old[i*4+3] = x1;
  }
  __syncthreads();

  for (int k = 0; k < NPOST; ++k) {
    float bs = -__builtin_inff(); int bi = 0;
    for (int i = tid; i < NPRE; i += 1024) {
      float s = live[i] ? tsc[i] : -__builtin_inff();
      if (s > bs) { bs = s; bi = i; }
    }
    rs[tid] = bs; ri[tid] = bi;
    __syncthreads();
    for (int off = 512; off > 0; off >>= 1) {
      if (tid < off) {
        float so = rs[tid + off]; int io = ri[tid + off];
        if (so > rs[tid] || (so == rs[tid] && io < ri[tid])) { rs[tid] = so; ri[tid] = io; }
      }
      __syncthreads();
    }
    if (tid == 0) {
      int i = ri[0]; float s = rs[0];
      int valid = (s > -__builtin_inff()) ? 1 : 0;
      keep_i[k] = valid ? i : 0;
      keep_v[k] = valid;
      bb[0] = cand[i*4+0]; bb[1] = cand[i*4+1];
      bb[2] = cand[i*4+2]; bb[3] = cand[i*4+3];
      bb[4] = areas[i];
    }
    __syncthreads();
    const float by0 = bb[0], bx0 = bb[1], by1 = bb[2], bx1 = bb[3], ba = bb[4];
    for (int i = tid; i < NPRE; i += 1024) {
      float yy0 = fmaxf(cand[i*4+0], by0);
      float xx0 = fmaxf(cand[i*4+1], bx0);
      float yy1 = fminf(cand[i*4+2], by1);
      float xx1 = fminf(cand[i*4+3], bx1);
      float inter = fmaxf(yy1 - yy0, 0.f) * fmaxf(xx1 - xx0, 0.f);
      float iou = inter / (areas[i] + ba - inter + 1e-9f);
      if (iou > 0.7f) live[i] = 0;
    }
    __syncthreads();
  }

  for (int k = tid; k < NPOST; k += 1024) {
    int v = keep_v[k]; int i = keep_i[k];
    out_rois[k*4+0] = v ? cand[i*4+0] : 0.f;
    out_rois[k*4+1] = v ? cand[i*4+1] : 0.f;
    out_rois[k*4+2] = v ? cand[i*4+2] : 0.f;
    out_rois[k*4+3] = v ? cand[i*4+3] : 0.f;
    out_scores[k] = v ? tsc[i] : 0.f;
    out_idx[k] = 0.0f;   // roi_indices are all-zero int32 (same bit pattern)
  }
}

// ---------------------------------------------------------------------------
extern "C" void kernel_launch(void* const* d_in, const int* in_sizes, int n_in,
                              void* d_out, int out_size, void* d_ws, size_t ws_size,
                              hipStream_t stream) {
  (void)in_sizes; (void)n_in; (void)out_size; (void)ws_size;
  const float* x       = (const float*)d_in[0];
  const int*   mask    = (const int*)  d_in[1];
  const float* conv1_w = (const float*)d_in[2];
  const float* conv1_b = (const float*)d_in[3];
  const float* loc_w   = (const float*)d_in[4];
  const float* loc_b   = (const float*)d_in[5];
  const float* cls_w   = (const float*)d_in[6];
  const float* cls_b   = (const float*)d_in[7];
  const float* cos_w   = (const float*)d_in[8];
  const float* cos_b   = (const float*)d_in[9];
  const int*   img_h   = (const int*)  d_in[10];
  const int*   img_w   = (const int*)  d_in[11];

  float* out = (float*)d_out;
  float* o_locs   = out;                 // 16384*4
  float* o_rois   = out + 65536;         // 300*4
  float* o_idx    = out + 66736;         // 300 (int32 zeros)
  float* o_anchor = out + 67036;         // 16384*4
  float* o_cls    = out + 132572;        // 16384*7
  float* o_cos    = out + 247260;        // 16384
  float* o_scores = out + 263644;        // 300
  float* o_old    = out + 263944;        // 6000*4

  // bump allocator over workspace (256B aligned)
  char* wsp = (char*)d_ws;
  size_t off = 0;
  auto alloc = [&](size_t bytes) -> void* {
    void* p = wsp + off;
    off += (bytes + 255) & ~(size_t)255;
    return p;
  };
  bf16*  xb      = (bf16*) alloc((size_t)HWPIX * CIN * sizeof(bf16));   // 16.8 MB
  bf16*  wbq     = (bf16*) alloc((size_t)9 * COUT * CIN * sizeof(bf16));// 4.7 MB
  float* hbuf    = (float*)alloc((size_t)HWPIX * COUT * sizeof(float)); // 33.6 MB
  float* ascale  = (float*)alloc(HWPIX * sizeof(float));
  float* acosA   = (float*)alloc(HWPIX * sizeof(float));
  float* fg      = (float*)alloc(HWPIX * sizeof(float));
  int*   keepm   = (int*)  alloc(HWPIX * sizeof(int));
  float* roi     = (float*)alloc((size_t)HWPIX * 4 * sizeof(float));
  float* scoresA = (float*)alloc(HWPIX * sizeof(float));
  float* cand    = (float*)alloc((size_t)NPRE * 4 * sizeof(float));
  float* tsc     = (float*)alloc(NPRE * sizeof(float));
  int*   live    = (int*)  alloc(NPRE * sizeof(int));
  float* areas   = (float*)alloc(NPRE * sizeof(float));

  transpose_x_kernel<<<(HWPIX * CIN) / 256, 256, 0, stream>>>(x, xb);
  prep_w_kernel<<<(9 * COUT * CIN) / 256, 256, 0, stream>>>(conv1_w, wbq);
  conv_wmma_kernel<<<HWPIX / MPIX, 256, SXELEM * sizeof(bf16), stream>>>(
      xb, wbq, conv1_b, mask, hbuf);
  heads_kernel<<<HWPIX / 256, 256, 0, stream>>>(hbuf, loc_w, loc_b, cls_w, cls_b,
                                                cos_w, cos_b, o_locs, o_cls, o_cos,
                                                fg, ascale, acosA, keepm);
  anchor_roi_kernel<<<HWPIX / 256, 256, 0, stream>>>(o_locs, fg, ascale, acosA, keepm,
                                                     img_h, img_w, o_anchor, roi, scoresA);
  sort_topk_kernel<<<1, 1024, HWPIX * 8, stream>>>(scoresA, roi, cand, tsc);
  nms_kernel<<<1, 1024, 0, stream>>>(cand, tsc, live, areas,
                                     o_rois, o_idx, o_scores, o_old);
}
